// MambaLayer_20753281975081
// MI455X (gfx1250) — matrix operand order repaired and checked by reading, not compiled, not run on hardware
//
#include <hip/hip_runtime.h>
#include <hip/hip_bf16.h>

// ---------------------------------------------------------------------------
// Types for CDNA5 WMMA
// ---------------------------------------------------------------------------
typedef _Float16 v16h __attribute__((ext_vector_type(16)));
typedef _Float16 h8   __attribute__((ext_vector_type(8)));
typedef _Float16 h4   __attribute__((ext_vector_type(4)));
typedef float    v8f  __attribute__((ext_vector_type(8)));
typedef int      v4i  __attribute__((vector_size(16)));

#if defined(__has_builtin)
# if __has_builtin(__builtin_amdgcn_global_load_async_to_lds_b128)
#  define USE_ASYNC_LDS 1
# endif
#endif
#ifndef USE_ASYNC_LDS
# define USE_ASYNC_LDS 0
#endif

__device__ __forceinline__ float silu_f(float v) {
  // fast silu: x * rcp(1+exp(-x)) -> v_exp_f32 + v_rcp_f32, no IEEE divide
  return v * __builtin_amdgcn_rcpf(1.0f + __expf(-v));
}

#if USE_ASYNC_LDS
__device__ __forceinline__ void async_cp16(_Float16* lds, const _Float16* g) {
  __builtin_amdgcn_global_load_async_to_lds_b128((v4i*)g, (v4i*)lds, 0, 0);
}
__device__ __forceinline__ void wait_async0() {
# if __has_builtin(__builtin_amdgcn_s_wait_asynccnt)
  __builtin_amdgcn_s_wait_asynccnt(0);
# else
  asm volatile("s_wait_asynccnt 0x0" ::: "memory");
# endif
}
#endif

// ---------------------------------------------------------------------------
// Operand pack: f32 (arbitrary strides, optional silu, two-level batch)
//   -> f16 canonical [z][i][k] row-major, zero-padded to (Ip, Kp)
// ---------------------------------------------------------------------------
__global__ __launch_bounds__(256)
void k_pack(const float* __restrict__ in, long long si, long long sk,
            long long szo, long long szi,
            _Float16* __restrict__ out, int I, int K, int Ip, int Kp,
            int binner, int act)
{
  long long idx = (long long)blockIdx.x * 256 + threadIdx.x;  // over Ip*Kp/4
  long long tot = (long long)Ip * Kp / 4;
  if (idx >= tot) return;
  int z  = blockIdx.z;
  int zo = z / binner, zi = z % binner;
  const float* ip = in + (long long)zo * szo + (long long)zi * szi;
  long long k0 = (idx * 4) % Kp;
  long long i  = (idx * 4) / Kp;
  h4 o;
#pragma unroll
  for (int u = 0; u < 4; ++u) {
    long long kk = k0 + u;
    float v = 0.0f;
    if (i < I && kk < K) v = ip[i * si + kk * sk];
    if (act) v = silu_f(v);
    o[u] = (_Float16)v;
  }
  *(h4*)(out + (long long)z * Ip * Kp + idx * 4) = o;
}

// ---------------------------------------------------------------------------
// Fragment build + 2x2 WMMA step from one LDS buffer
// ---------------------------------------------------------------------------
__device__ __forceinline__ void frag_mma(const _Float16 (*__restrict__ As)[32],
                                         const _Float16 (*__restrict__ Bs)[32],
                                         int wm, int wn, int lane, v8f acc[2][2])
{
  const int hl    = lane >> 4;        // 0 | 1
  const int l16   = lane & 15;
  const int abase = hl ? 8 : 0;       // A frag: i<8 -> K=abase+i ; i>=8 -> K=16+abase+(i-8)
  const int bbase = hl ? 8 : 0;       // B frag: i<8 -> K=bbase+i ; i>=8 -> K=16+bbase+(i-8)

  v16h afr[2], bfr[2];
#pragma unroll
  for (int mi = 0; mi < 2; ++mi) {
    int m = wm * 32 + mi * 16 + l16;
    union { v16h v; h8 h[2]; } u;
    u.h[0] = *(const h8*)&As[m][abase];
    u.h[1] = *(const h8*)&As[m][16 + abase];
    afr[mi] = u.v;
  }
#pragma unroll
  for (int ni = 0; ni < 2; ++ni) {
    int n = wn * 32 + ni * 16 + l16;
    union { v16h v; h8 h[2]; } u;
    u.h[0] = *(const h8*)&Bs[n][bbase];
    u.h[1] = *(const h8*)&Bs[n][bbase + 16];
    bfr[ni] = u.v;
  }
#pragma unroll
  for (int mi = 0; mi < 2; ++mi)
#pragma unroll
    for (int ni = 0; ni < 2; ++ni)
      acc[mi][ni] = __builtin_amdgcn_wmma_f32_16x16x32_f16(
          false, afr[mi], false, bfr[ni], (short)0, acc[mi][ni], false, false);
}

// ---------------------------------------------------------------------------
// f16-operand batched GEMM:  C[i,j] = sum_k A16[z][i][k]*B16[z][j][k] (+biasI)(+Cin)
// A16: (Mp x Kp) row-major per batch; B16: (Np x Kp) row-major per batch (B^T).
// Tile 64(M) x 128(N), K-step 32, 8 waves (2x4), 2x2 WMMA 16x16x32 per wave.
// Double-buffered async global->LDS staging: next K-step's copies are issued
// right after the barrier and land while the current step's WMMAs execute.
// ---------------------------------------------------------------------------
__global__ __launch_bounds__(256)
void k_gemm16(const _Float16* __restrict__ A16, long long aB,
              const _Float16* __restrict__ B16, long long bB,
              const float* __restrict__ Cin, const float* __restrict__ biasI,
              float* __restrict__ Cout, long long csi, long long csj,
              long long sCo, long long sCi,
              int M, int N, int Kp, int binner)
{
  __shared__ _Float16 As[2][64][32];
  __shared__ _Float16 Bs[2][128][32];

  const int tid  = threadIdx.x;
  const int lane = tid & 31;
  const int wv   = tid >> 5;
  const int wm   = wv >> 2;           // 0..1
  const int wn   = wv & 3;            // 0..3
  const int n0   = blockIdx.x * 128;
  const int m0   = blockIdx.y * 64;
  const int z    = blockIdx.z;
  const int zo   = z / binner, zi = z % binner;

  const _Float16* Ab = A16 + (long long)z * aB + (long long)m0 * Kp;
  const _Float16* Bb = B16 + (long long)z * bB + (long long)n0 * Kp;
  const long long coff = (long long)zo * sCo + (long long)zi * sCi;

  // staging mapping: each lane copies 16B (8 halves)
  const int srow = lane >> 2;           // 0..7
  const int sc8  = (lane & 3) * 8;      // 0,8,16,24
  const int ra   = wv * 8 + srow;       // A row this lane stages (64 rows)
  const int rb   = wv * 16 + srow;      // B rows this lane stages (+0 / +8)

  v8f acc[2][2];
#pragma unroll
  for (int mi = 0; mi < 2; ++mi)
#pragma unroll
    for (int ni = 0; ni < 2; ++ni)
#pragma unroll
      for (int r = 0; r < 8; ++r) acc[mi][ni][r] = 0.0f;

#if USE_ASYNC_LDS
  // prologue: issue K-step 0 into buffer 0
  async_cp16(&As[0][ra][sc8],     Ab + (long long)ra * Kp + sc8);
  async_cp16(&Bs[0][rb][sc8],     Bb + (long long)rb * Kp + sc8);
  async_cp16(&Bs[0][rb + 8][sc8], Bb + (long long)(rb + 8) * Kp + sc8);
  int cb = 0;
  for (int kt = 0; kt < Kp; kt += 32, cb ^= 1) {
    wait_async0();          // current buffer's copies (issued an iteration ago) done
    __syncthreads();        // all waves' copies visible; all reads of other buffer done
    if (kt + 32 < Kp) {
      const int nb = cb ^ 1;
      const int kn = kt + 32;
      async_cp16(&As[nb][ra][sc8],     Ab + (long long)ra * Kp + kn + sc8);
      async_cp16(&Bs[nb][rb][sc8],     Bb + (long long)rb * Kp + kn + sc8);
      async_cp16(&Bs[nb][rb + 8][sc8], Bb + (long long)(rb + 8) * Kp + kn + sc8);
    }
    frag_mma(As[cb], Bs[cb], wm, wn, lane, acc);
  }
#else
  for (int kt = 0; kt < Kp; kt += 32) {
    __syncthreads();
    *(h8*)&As[0][ra][sc8]     = *(const h8*)(Ab + (long long)ra * Kp + kt + sc8);
    *(h8*)&Bs[0][rb][sc8]     = *(const h8*)(Bb + (long long)rb * Kp + kt + sc8);
    *(h8*)&Bs[0][rb + 8][sc8] = *(const h8*)(Bb + (long long)(rb + 8) * Kp + kt + sc8);
    __syncthreads();
    frag_mma(As[0], Bs[0], wm, wn, lane, acc);
  }
#endif

  // epilogue: C layout: VGPR r -> row r + 8*(lane>=16), col = lane&15
  const int l16 = lane & 15;
  const int rh  = (lane >> 4) * 8;
#pragma unroll
  for (int mi = 0; mi < 2; ++mi)
#pragma unroll
    for (int ni = 0; ni < 2; ++ni)
#pragma unroll
      for (int r = 0; r < 8; ++r) {
        int gi = m0 + wm * 32 + mi * 16 + rh + r;
        int gj = n0 + wn * 32 + ni * 16 + l16;
        if (gi < M && gj < N) {
          float v = acc[mi][ni][r];
          if (biasI) v += biasI[gi];
          long long idx = coff + (long long)gi * csi + (long long)gj * csj;
          if (Cin) v += Cin[idx];
          Cout[idx] = v;
        }
      }
}

// ---------------------------------------------------------------------------
// LayerNorm over C=256 (one block per row)
// ---------------------------------------------------------------------------
__global__ __launch_bounds__(256)
void k_layernorm(const float* __restrict__ x, const float* __restrict__ g,
                 const float* __restrict__ b, float* __restrict__ y, int R)
{
  __shared__ float red[256];
  int r = blockIdx.x;
  int t = threadIdx.x;
  float v = x[(long long)r * 256 + t];
  red[t] = v; __syncthreads();
  for (int s = 128; s > 0; s >>= 1) { if (t < s) red[t] += red[t + s]; __syncthreads(); }
  float mean = red[0] / 256.0f;
  __syncthreads();
  float d = v - mean;
  red[t] = d * d; __syncthreads();
  for (int s = 128; s > 0; s >>= 1) { if (t < s) red[t] += red[t + s]; __syncthreads(); }
  float var = red[0] / 256.0f;
  y[(long long)r * 256 + t] = d * rsqrtf(var + 1e-5f) * g[t] + b[t];
}

// ---------------------------------------------------------------------------
// Permutes (B,C,N) <-> (B,N,C);  C=256, N=1024
// ---------------------------------------------------------------------------
__global__ __launch_bounds__(256)
void k_perm_in(const float* __restrict__ in, float* __restrict__ out)
{
  long long i = (long long)blockIdx.x * 256 + threadIdx.x;   // over B*C*N
  long long n = i & 1023;
  long long c = (i >> 10) & 255;
  long long b = i >> 18;
  out[b * 262144 + n * 256 + c] = in[i];
}

__global__ __launch_bounds__(256)
void k_perm_out(const float* __restrict__ in, float* __restrict__ out)
{
  long long i = (long long)blockIdx.x * 256 + threadIdx.x;   // over B*C*N
  long long n = i & 1023;
  long long c = (i >> 10) & 255;
  long long b = i >> 18;
  out[i] = in[b * 262144 + n * 256 + c];
}

// ---------------------------------------------------------------------------
// Depthwise 3x3 conv, SAME, on (B, 768, 32, 32)
// ---------------------------------------------------------------------------
__global__ __launch_bounds__(256)
void k_dwconv2d(const float* __restrict__ in, const float* __restrict__ w,
                const float* __restrict__ bias, float* __restrict__ out)
{
  long long i = (long long)blockIdx.x * 256 + threadIdx.x;   // B*768*1024
  int px = (int)(i & 31);
  int py = (int)((i >> 5) & 31);
  long long bc = i >> 10;            // 0..3071
  int oc = (int)(bc % 768);
  float acc = bias[oc];
#pragma unroll
  for (int dy = 0; dy < 3; ++dy)
#pragma unroll
    for (int dx = 0; dx < 3; ++dx) {
      int yy = py + dy - 1, xx = px + dx - 1;
      if (yy >= 0 && yy < 32 && xx >= 0 && xx < 32)
        acc += in[bc * 1024 + yy * 32 + xx] * w[oc * 9 + dy * 3 + dx];
    }
  out[i] = acc;
}

// ---------------------------------------------------------------------------
// L2-normalize q and k rows (length 1024) in-place in qkv2 (b, 768, 1024)
// ---------------------------------------------------------------------------
__global__ __launch_bounds__(256)
void k_qknorm(float* __restrict__ qkv2)
{
  __shared__ float red[256];
  int row = blockIdx.x;
  int b = row >> 9, ch = row & 511;
  float* p = qkv2 + (long long)b * 786432 + (long long)ch * 1024;
  int t = threadIdx.x;
  float vals[4]; float ss = 0.0f;
#pragma unroll
  for (int u = 0; u < 4; ++u) { vals[u] = p[t + 256 * u]; ss += vals[u] * vals[u]; }
  red[t] = ss; __syncthreads();
  for (int s = 128; s > 0; s >>= 1) { if (t < s) red[t] += red[t + s]; __syncthreads(); }
  float sc = 1.0f / fmaxf(sqrtf(red[0]), 1e-12f);
#pragma unroll
  for (int u = 0; u < 4; ++u) p[t + 256 * u] = vals[u] * sc;
}

// ---------------------------------------------------------------------------
// Scale by temp[h] + softmax over rows of 64 (att: b,h,64,64)
// ---------------------------------------------------------------------------
__global__ __launch_bounds__(64)
void k_softmax(float* __restrict__ att, const float* __restrict__ temp)
{
  __shared__ float red[64];
  int row = blockIdx.x;               // b*256 + h*64 + c
  int h = (row >> 6) & 3;
  float* p = att + (long long)row * 64;
  int t = threadIdx.x;
  float v = p[t] * temp[h];
  red[t] = v; __syncthreads();
  for (int s = 32; s > 0; s >>= 1) { if (t < s) red[t] = fmaxf(red[t], red[t + s]); __syncthreads(); }
  float mx = red[0]; __syncthreads();
  float e = __expf(v - mx);
  red[t] = e; __syncthreads();
  for (int s = 32; s > 0; s >>= 1) { if (t < s) red[t] += red[t + s]; __syncthreads(); }
  p[t] = e * __builtin_amdgcn_rcpf(red[0]);
}

// ---------------------------------------------------------------------------
// Depthwise 1D conv (k=3, SAME) + SiLU.  128 channels, L=1024.
// ---------------------------------------------------------------------------
__global__ __launch_bounds__(256)
void k_dwconv1d_silu(const float* __restrict__ in, int inCtot, int in_choff,
                     const float* __restrict__ w,
                     float* __restrict__ out, int outCtot, int out_choff)
{
  long long i = (long long)blockIdx.x * 256 + threadIdx.x;  // B*128*1024
  int l  = (int)(i & 1023);
  int dc = (int)((i >> 10) & 127);
  int b  = (int)(i >> 17);
  const float* ip = in + ((long long)b * inCtot + in_choff + dc) * 1024;
  float acc = 0.0f;
#pragma unroll
  for (int t = 0; t < 3; ++t) {
    int ll = l + t - 1;
    if (ll >= 0 && ll < 1024) acc += ip[ll] * w[dc * 3 + t];
  }
  out[((long long)b * outCtot + out_choff + dc) * 1024 + l] = silu_f(acc);
}

// ---------------------------------------------------------------------------
// Cubic B-spline bases (12 uniform knots: -2.2 + 0.4*j), 8 outputs per value.
// in: (R, Cd) f32 ; out f16 packed: out[(r*Cd+c)*8 + t]
// ---------------------------------------------------------------------------
__global__ __launch_bounds__(256)
void k_bsplines16(const float* __restrict__ x, _Float16* __restrict__ out, long long total)
{
  long long i = (long long)blockIdx.x * 256 + threadIdx.x;
  if (i >= total) return;
  float xv = x[i];
  const float t0 = -2.2f, hstep = 0.4f;
  float bas[11];
#pragma unroll
  for (int j = 0; j < 11; ++j) {
    float tj = t0 + hstep * j;
    bas[j] = (xv >= tj && xv < tj + hstep) ? 1.0f : 0.0f;
  }
#pragma unroll
  for (int k = 1; k <= 3; ++k) {
    float inv = 1.0f / (hstep * k);
#pragma unroll
    for (int j = 0; j < 10; ++j) {
      if (j < 11 - k) {
        float tj = t0 + hstep * j;
        float lft = (xv - tj) * inv * bas[j];
        float rgt = ((tj + hstep * (k + 1)) - xv) * inv * bas[j + 1];
        bas[j] = lft + rgt;
      }
    }
  }
  h8 o;
#pragma unroll
  for (int t = 0; t < 8; ++t) o[t] = (_Float16)bas[t];
  *(h8*)(out + i * 8) = o;
}

// scaled16[o, c*8+t] = sw[o,c,t] * ss[o,c]   (f16 packed B operand)
__global__ __launch_bounds__(256)
void k_make_scaled16(const float* __restrict__ sw, const float* __restrict__ ss,
                     _Float16* __restrict__ out, long long totalOC)
{
  long long i = (long long)blockIdx.x * 256 + threadIdx.x;   // over O*Cd
  if (i >= totalOC) return;
  float s = ss[i];
  const float* sp = sw + i * 8;
  h8 r;
#pragma unroll
  for (int t = 0; t < 8; ++t) r[t] = (_Float16)(sp[t] * s);
  *(h8*)(out + i * 8) = r;
}

// ---------------------------------------------------------------------------
// Selective scan: B=4, D=128, L=1024, Nstate=8, sequential over L.
// ---------------------------------------------------------------------------
__global__ __launch_bounds__(256)
void k_scan(const float* __restrict__ dt, const float* __restrict__ xm,
            const float* __restrict__ xdbl, const float* __restrict__ A_log,
            const float* __restrict__ Dp, const float* __restrict__ dtb,
            float* __restrict__ ycat)
{
  int t = blockIdx.x * 256 + threadIdx.x;
  if (t >= 512) return;
  int b = t >> 7, d = t & 127;
  float Ar[8], h[8];
#pragma unroll
  for (int n = 0; n < 8; ++n) { Ar[n] = -__expf(A_log[d * 8 + n]); h[n] = 0.0f; }
  const float* dtp = dt + ((long long)b * 128 + d) * 1024;
  const float* xp  = xm + ((long long)b * 128 + d) * 1024;
  const float* xd  = xdbl + (long long)b * 32768;
  float* yp = ycat + ((long long)b * 256 + d) * 1024;
  float db = dtb[d], Dv = Dp[d];
  for (int l = 0; l < 1024; ++l) {
    float z = dtp[l] + db;
    float delta = (z > 20.0f) ? z : log1pf(__expf(z));
    float xv = xp[l];
    float dx = delta * xv;
    float y = 0.0f;
    const float* row = xd + (long long)l * 32;
#pragma unroll
    for (int n = 0; n < 8; ++n) {
      h[n] = __expf(delta * Ar[n]) * h[n] + dx * row[16 + n];
      y += h[n] * row[24 + n];
    }
    yp[l] = y + Dv * xv;
  }
}

// ---------------------------------------------------------------------------
// Host-side orchestration
// ---------------------------------------------------------------------------
static inline void pack(hipStream_t st, const float* in, long long si, long long sk,
                        long long szo, long long szi, _Float16* out,
                        int I, int K, int Ip, int Kp, int bo, int bi, int act)
{
  long long tot = (long long)Ip * Kp / 4;
  dim3 g((unsigned)((tot + 255) / 256), 1, bo * bi);
  k_pack<<<g, 256, 0, st>>>(in, si, sk, szo, szi, out, I, K, Ip, Kp, bi, act);
}

static inline void gemm16(hipStream_t st, const _Float16* A, long long aB,
                          const _Float16* B, long long bB,
                          const float* Cin, const float* biasI, float* C,
                          long long csi, long long csj, long long sCo, long long sCi,
                          int M, int N, int Kp, int bo, int bi)
{
  dim3 g((N + 127) / 128, (M + 63) / 64, bo * bi);
  k_gemm16<<<g, 256, 0, st>>>(A, aB, B, bB, Cin, biasI, C, csi, csj, sCo, sCi,
                              M, N, Kp, bi);
}

static void run_fkan(hipStream_t st,
                     const float* ln_g, const float* ln_b,
                     const float* bw1, const float* sw1, const float* ss1,
                     const float* bw2, const float* sw2, const float* ss2,
                     float* X, float* TMP, float* H, _Float16* PA, _Float16* PB)
{
  k_layernorm<<<4096, 256, 0, st>>>(X, ln_g, ln_b, TMP, 4096);
  // base1: H = silu(TMP) @ bw1^T    (M=4096, K=256, N=1024)
  pack(st, TMP, 256, 1, 0, 0, PA, 4096, 256, 4096, 256, 1, 1, 1);
  pack(st, bw1, 256, 1, 0, 0, PB, 1024, 256, 1024, 256, 1, 1, 0);
  gemm16(st, PA, 0, PB, 0, nullptr, nullptr, H, 1024, 1, 0, 0, 4096, 1024, 256, 1, 1);
  // spline1: H += bsplines(TMP) @ scaled1^T  (K=2048)
  k_bsplines16<<<(1048576 + 255) / 256, 256, 0, st>>>(TMP, PA, 1048576LL);
  k_make_scaled16<<<(262144 + 255) / 256, 256, 0, st>>>(sw1, ss1, PB, 262144LL);
  gemm16(st, PA, 0, PB, 0, H, nullptr, H, 1024, 1, 0, 0, 4096, 1024, 2048, 1, 1);
  // base2 + residual: TMP = silu(H) @ bw2^T + X  (K=1024)
  pack(st, H, 1024, 1, 0, 0, PA, 4096, 1024, 4096, 1024, 1, 1, 1);
  pack(st, bw2, 1024, 1, 0, 0, PB, 256, 1024, 256, 1024, 1, 1, 0);
  gemm16(st, PA, 0, PB, 0, X, nullptr, TMP, 256, 1, 0, 0, 4096, 256, 1024, 1, 1);
  // spline2 chunked (rows of 1024): X = TMP + bsplines(H) @ scaled2^T  (K=8192)
  k_make_scaled16<<<(262144 + 255) / 256, 256, 0, st>>>(sw2, ss2, PB, 262144LL);
  for (int c = 0; c < 4; ++c) {
    long long R0 = (long long)c * 1024;
    k_bsplines16<<<(1048576 + 255) / 256, 256, 0, st>>>(H + R0 * 1024, PA, 1048576LL);
    gemm16(st, PA, 0, PB, 0, TMP + R0 * 256, nullptr, X + R0 * 256,
           256, 1, 0, 0, 1024, 256, 8192, 1, 1);
  }
}

extern "C" void kernel_launch(void* const* d_in, const int* in_sizes, int n_in,
                              void* d_out, int out_size, void* d_ws, size_t ws_size,
                              hipStream_t stream)
{
  (void)in_sizes; (void)n_in; (void)out_size; (void)ws_size;

  const float* x5d    = (const float*)d_in[0];
  const float* ln1_g  = (const float*)d_in[1];
  const float* ln1_b  = (const float*)d_in[2];
  const float* temp   = (const float*)d_in[3];
  const float* qkv_w  = (const float*)d_in[4];
  const float* qkv_b  = (const float*)d_in[5];
  const float* dw_w   = (const float*)d_in[6];
  const float* dw_b   = (const float*)d_in[7];
  const float* po_w   = (const float*)d_in[8];
  const float* po_b   = (const float*)d_in[9];
  const float* f_ln_g[2] = {(const float*)d_in[10], (const float*)d_in[18]};
  const float* f_ln_b[2] = {(const float*)d_in[11], (const float*)d_in[19]};
  const float* f_bw1[2]  = {(const float*)d_in[12], (const float*)d_in[20]};
  const float* f_sw1[2]  = {(const float*)d_in[13], (const float*)d_in[21]};
  const float* f_ss1[2]  = {(const float*)d_in[14], (const float*)d_in[22]};
  const float* f_bw2[2]  = {(const float*)d_in[15], (const float*)d_in[23]};
  const float* f_sw2[2]  = {(const float*)d_in[16], (const float*)d_in[24]};
  const float* f_ss2[2]  = {(const float*)d_in[17], (const float*)d_in[25]};
  const float* ln3_g  = (const float*)d_in[26];
  const float* ln3_b  = (const float*)d_in[27];
  const float* in_w   = (const float*)d_in[28];
  const float* cx_w   = (const float*)d_in[29];
  const float* cz_w   = (const float*)d_in[30];
  const float* xp_w   = (const float*)d_in[31];
  const float* dtp_w  = (const float*)d_in[32];
  const float* dtp_b  = (const float*)d_in[33];
  const float* A_log  = (const float*)d_in[34];
  const float* Dp     = (const float*)d_in[35];
  const float* out_w  = (const float*)d_in[36];

  // workspace carve-up
  char* wp = (char*)d_ws;
  auto alloc = [&](size_t bytes) -> char* {
    char* p = wp;
    wp += (bytes + 255) & ~(size_t)255;
    return p;
  };
  float* X        = (float*)alloc(4194304);     // (B,N,C) residual
  float* TMP      = (float*)alloc(4194304);     // (B,N,C) scratch
  float* QKV      = (float*)alloc(12582912);    // (B,768,1024)
  float* QKV2     = (float*)alloc(12582912);    // (B,768,1024)
  float* ATT      = (float*)alloc(262144);      // (B,H,64,64)
  float* ATTO     = (float*)alloc(4194304);     // (B,256,1024)
  float* H        = (float*)alloc(16777216);    // (4096,1024)
  _Float16* PA    = (_Float16*)alloc(16777216); // packed A operand (f16)
  _Float16* PB    = (_Float16*)alloc(4194304);  // packed B operand (f16)
  float* XZ       = (float*)alloc(4194304);     // (B,256,1024)
  float* XM       = (float*)alloc(2097152);     // (B,128,1024)
  float* XDBL     = (float*)alloc(524288);      // (B,1024,32)
  float* DTb      = (float*)alloc(2097152);     // (B,128,1024)
  float* YCAT     = (float*)alloc(4194304);     // (B,256,1024)

  // ---- input permute (B,C,N) -> (B,N,C)
  k_perm_in<<<4096, 256, 0, stream>>>(x5d, X);

  // ================= MDTA =================
  k_layernorm<<<4096, 256, 0, stream>>>(X, ln1_g, ln1_b, TMP, 4096);
  // qkv 1x1: QKV[b,oc,n] = qkv_w @ TMP^T + qkv_b   (M=768, K=256, N=1024)
  pack(stream, qkv_w, 256, 1, 0, 0, PA, 768, 256, 768, 256, 1, 1, 0);
  pack(stream, TMP, 256, 1, 262144, 0, PB, 1024, 256, 1024, 256, 4, 1, 0);
  gemm16(stream, PA, 0, PB, 262144, nullptr, qkv_b, QKV,
         1024, 1, 786432, 0, 768, 1024, 256, 4, 1);
  k_dwconv2d<<<12288, 256, 0, stream>>>(QKV, dw_w, dw_b, QKV2);
  k_qknorm<<<2048, 256, 0, stream>>>(QKV2);
  // attn = q @ k^T per (b,h): M=64, K=1024, N=64
  pack(stream, QKV2,          1024, 1, 786432, 65536, PA, 64, 1024, 64, 1024, 4, 4, 0);
  pack(stream, QKV2 + 262144, 1024, 1, 786432, 65536, PB, 64, 1024, 128, 1024, 4, 4, 0);
  gemm16(stream, PA, 65536, PB, 131072, nullptr, nullptr, ATT,
         64, 1, 16384, 4096, 64, 64, 1024, 4, 4);
  k_softmax<<<1024, 64, 0, stream>>>(ATT, temp);
  // out = attn @ v per (b,h): M=64, K=64, N=1024 -> ATTO (b,256,1024)
  pack(stream, ATT, 64, 1, 16384, 4096, PA, 64, 64, 64, 64, 4, 4, 0);
  pack(stream, QKV2 + 524288, 1, 1024, 786432, 65536, PB, 1024, 64, 1024, 64, 4, 4, 0);
  gemm16(stream, PA, 4096, PB, 65536, nullptr, nullptr, ATTO,
         1024, 1, 262144, 65536, 64, 1024, 64, 4, 4);
  // project + residual: X[b,n,co] = po_w @ ATTO + po_b + X
  pack(stream, po_w, 256, 1, 0, 0, PA, 256, 256, 256, 256, 1, 1, 0);
  pack(stream, ATTO, 1, 1024, 262144, 0, PB, 1024, 256, 1024, 256, 4, 1, 0);
  gemm16(stream, PA, 0, PB, 262144, X, po_b, X,
         1, 256, 262144, 0, 256, 1024, 256, 4, 1);

  // ================= FKAN block 1 =================
  run_fkan(stream, f_ln_g[0], f_ln_b[0], f_bw1[0], f_sw1[0], f_ss1[0],
           f_bw2[0], f_sw2[0], f_ss2[0], X, TMP, H, PA, PB);

  // ================= Mamba mixer =================
  k_layernorm<<<4096, 256, 0, stream>>>(X, ln3_g, ln3_b, TMP, 4096);
  // xz[b,d,l] = in_w @ TMP^T   (M=256, K=256, N=1024)
  pack(stream, in_w, 256, 1, 0, 0, PA, 256, 256, 256, 256, 1, 1, 0);
  pack(stream, TMP, 256, 1, 262144, 0, PB, 1024, 256, 1024, 256, 4, 1, 0);
  gemm16(stream, PA, 0, PB, 262144, nullptr, nullptr, XZ,
         1024, 1, 262144, 0, 256, 1024, 256, 4, 1);
  k_dwconv1d_silu<<<2048, 256, 0, stream>>>(XZ, 256, 0,   cx_w, XM,   128, 0);
  k_dwconv1d_silu<<<2048, 256, 0, stream>>>(XZ, 256, 128, cz_w, YCAT, 256, 128);
  // x_dbl[b,l,e] = xm^T @ xp_w^T   (M=1024, K=128, N=32)
  pack(stream, XM, 1, 1024, 131072, 0, PA, 1024, 128, 1024, 128, 4, 1, 0);
  pack(stream, xp_w, 128, 1, 0, 0, PB, 32, 128, 128, 128, 1, 1, 0);
  gemm16(stream, PA, 131072, PB, 0, nullptr, nullptr, XDBL,
         32, 1, 32768, 0, 1024, 32, 128, 4, 1);
  // dt[b,d,l] = dtp_w @ x_dbl[...,:16]^T   (M=128, K=16->32, N=1024)
  pack(stream, dtp_w, 16, 1, 0, 0, PA, 128, 16, 128, 32, 1, 1, 0);
  pack(stream, XDBL, 32, 1, 32768, 0, PB, 1024, 16, 1024, 32, 4, 1, 0);
  gemm16(stream, PA, 0, PB, 32768, nullptr, nullptr, DTb,
         1024, 1, 131072, 0, 128, 1024, 32, 4, 1);
  // selective scan -> YCAT ch 0..127
  k_scan<<<2, 256, 0, stream>>>(DTb, XM, XDBL, A_log, Dp, dtp_b, YCAT);
  // out projection + residual: X[b,l,c] = YCAT^T @ out_w^T + X
  pack(stream, YCAT, 1, 1024, 262144, 0, PA, 1024, 256, 1024, 256, 4, 1, 0);
  pack(stream, out_w, 256, 1, 0, 0, PB, 256, 256, 256, 256, 1, 1, 0);
  gemm16(stream, PA, 262144, PB, 0, X, nullptr, X,
         256, 1, 262144, 0, 1024, 256, 256, 4, 1);

  // ================= FKAN block 2 =================
  run_fkan(stream, f_ln_g[1], f_ln_b[1], f_bw1[1], f_sw1[1], f_ss1[1],
           f_bw2[1], f_sw2[1], f_ss2[1], X, TMP, H, PA, PB);

  // ---- output permute (B,N,C) -> (B,C,N)
  k_perm_out<<<4096, 256, 0, stream>>>(X, (float*)d_out);
}
